// BuildSubGraph_32615981645853
// MI455X (gfx1250) — matrix operand order repaired and checked by reading, not compiled, not run on hardware
//
#include <hip/hip_runtime.h>

#define NCAT 3000
#define NP   3008          // padded to multiple of 16/32
#define NT   (NP/16)       // 188 row/col tiles
#define HID  64
#define NG   4
#define NL   2
#define HOPS 4
#define ALPHAC 0.15f
#define EPSV 1e-5f
#define BB 256
#define LL 50

typedef _Float16 f16;
typedef __attribute__((ext_vector_type(16))) _Float16 v16h;
typedef __attribute__((ext_vector_type(8)))  _Float16 v8h;
typedef __attribute__((ext_vector_type(8)))  float    v8f;
typedef int v4i_vec __attribute__((vector_size(4 * sizeof(int))));

#if __has_builtin(__builtin_amdgcn_global_load_async_to_lds_b128)
#define HAS_ASYNC_LDS 1
#else
#define HAS_ASYNC_LDS 0
#endif

#define WMMA_F32_F16(a,b,c) \
  __builtin_amdgcn_wmma_f32_16x16x32_f16(false,(a),false,(b),(short)0,(c),false,false)

// A operand (16x32 f16) from row-major matrix, leading dim ld (halves).
// ISA layout: lane = row (m=lane%16); half group 0: K 0-7 then 16-23; group 1: K 8-15 then 24-31.
__device__ __forceinline__ v16h frag_a_rm(const f16* base, int ld,
                                          int row0, int k0, int lane) {
  const int m = lane & 15, half = lane >> 4;
  const f16* p = base + (size_t)(row0 + m) * ld + (k0 + 8 * half);
  v8h lo = *(const v8h*)p;          // K = 8*half + 0..7
  v8h hi = *(const v8h*)(p + 16);   // K = 8*half + 16..23
  v16h r;
#pragma unroll
  for (int j = 0; j < 8; ++j) { r[j] = lo[j]; r[j + 8] = hi[j]; }
  return r;
}

// B operand (32x16 f16). ISA layout: lane = out-column n (lane%16); half group 0: K 0-7 then 16-23; group 1: K 8-15 then 24-31.
// Source stored so element (n,k) lives at base[(n0+n)*ld + k0 + k].
__device__ __forceinline__ v16h frag_b_cm(const f16* __restrict__ base, int ld,
                                          int n0, int k0, int lane) {
  const int n = lane & 15, half = lane >> 4;
  const f16* p = base + (size_t)(n0 + n) * ld + (k0 + 8 * half);
  v8h lo = *(const v8h*)p;
  v8h hi = *(const v8h*)(p + 16);
  v16h r;
#pragma unroll
  for (int j = 0; j < 8; ++j) { r[j] = lo[j]; r[j + 8] = hi[j]; }
  return r;
}

#if HAS_ASYNC_LDS
// Async-copy one 16B chunk global -> LDS (GLOBAL_LOAD_ASYNC_TO_LDS_B128).
// Builtin takes int4 pointers: AS(1) global (raw VA) and AS(3) LDS (generic LDS
// pointers carry the LDS byte offset in their low 32 bits per the aperture rule).
typedef __attribute__((address_space(1))) v4i_vec* g_v4p;
typedef __attribute__((address_space(3))) v4i_vec* l_v4p;
__device__ __forceinline__ void async_g2l_b128(const f16* g, void* lds) {
  __builtin_amdgcn_global_load_async_to_lds_b128(
      (g_v4p)(unsigned long long)g,
      (l_v4p)(unsigned)(unsigned long long)lds,
      0, 0);
}
#endif

// ---- padded h (f32 + f16) from emb; zero the tiny accumulator buffer ----
__global__ void k_init(const float* __restrict__ emb, float* __restrict__ hf32,
                       f16* __restrict__ hf16, float* __restrict__ sgc) {
  int i = blockIdx.x * 256 + threadIdx.x;
  if (i < NP * HID) {
    int n = i / HID;
    float v = (n < NCAT) ? emb[i] : 0.f;
    hf32[i] = v;
    hf16[i] = (f16)v;
  }
  if (i < NG * HID + 1) sgc[i] = 0.f;   // S[g][c] and Wp-sum slot
}

// ---- cast Wq/Wk/Wv to f16, transposed to [l][out][in] for contiguous B frags ----
__global__ void k_castw(const float* __restrict__ Wq, const float* __restrict__ Wk,
                        const float* __restrict__ Wv, f16* __restrict__ wqT,
                        f16* __restrict__ wkT, f16* __restrict__ wvT) {
  int t = blockIdx.x * 256 + threadIdx.x;
  if (t >= NL * HID * HID) return;
  int l = t / (HID * HID), r = t % (HID * HID);
  int i = r / HID, o = r % HID;        // W[l][i][o]
  int dst = l * HID * HID + o * HID + i;
  wqT[dst] = (f16)Wq[t];
  wkT[dst] = (f16)Wk[t];
  wvT[dst] = (f16)Wv[t];
}

// ---- q,k,v = h @ W : wave per (row-tile, col-tile), 3 accumulators share A frag ----
__global__ void k_qkv(const f16* __restrict__ hf16, const f16* __restrict__ wqT,
                      const f16* __restrict__ wkT, const f16* __restrict__ wvT,
                      f16* __restrict__ qf, f16* __restrict__ kf,
                      float* __restrict__ vf, f16* __restrict__ vC) {
  int wave = (blockIdx.x * blockDim.x + threadIdx.x) >> 5;
  int lane = threadIdx.x & 31;
  if (wave >= NT * 4) return;
  int rt = wave >> 2, ct = wave & 3;
  v8f aq = {}, ak = {}, av = {};
#pragma unroll
  for (int k0 = 0; k0 < HID; k0 += 32) {
    v16h a  = frag_a_rm(hf16, HID, rt * 16, k0, lane);
    v16h bq = frag_b_cm(wqT, HID, ct * 16, k0, lane);
    v16h bk = frag_b_cm(wkT, HID, ct * 16, k0, lane);
    v16h bv = frag_b_cm(wvT, HID, ct * 16, k0, lane);
    aq = WMMA_F32_F16(a, bq, aq);
    ak = WMMA_F32_F16(a, bk, ak);
    av = WMMA_F32_F16(a, bv, av);
  }
  int half = lane >> 4, c = ct * 16 + (lane & 15);
  v8h vpack;
#pragma unroll
  for (int r = 0; r < 8; ++r) {                 // D: M = r + 8*half, N = lane%16
    int m = rt * 16 + 8 * half + r;
    int idx = m * HID + c;
    qf[idx] = (f16)aq[r];
    kf[idx] = (f16)ak[r];
    vf[idx] = av[r];
    vpack[r] = (f16)av[r];
  }
  *(v8h*)(vC + (size_t)c * NP + rt * 16 + 8 * half) = vpack;  // column-major v (= z0)
}

// ---- scores = (q @ k^T)/8, masked by adj ----
__global__ void k_scores(const f16* __restrict__ qf, const f16* __restrict__ kf,
                         const float* __restrict__ adj, float* __restrict__ Sb) {
  int wave = (blockIdx.x * blockDim.x + threadIdx.x) >> 5;
  int lane = threadIdx.x & 31;
  if (wave >= NT * NT) return;
  int rt = wave / NT, ctile = wave % NT;
  v8f acc = {};
#pragma unroll
  for (int k0 = 0; k0 < HID; k0 += 32) {
    v16h a = frag_a_rm(qf, HID, rt * 16, k0, lane);
    v16h b = frag_b_cm(kf, HID, ctile * 16, k0, lane);   // B[k][n] = k[n][k]
    acc = WMMA_F32_F16(a, b, acc);
  }
  int half = lane >> 4, n = ctile * 16 + (lane & 15);
#pragma unroll
  for (int r = 0; r < 8; ++r) {
    int m = rt * 16 + 8 * half + r;
    float v = acc[r] * 0.125f;                  // 1/sqrt(64)
    if (m < NCAT && n < NCAT)
      v = (adj[(size_t)m * NCAT + n] > 0.f) ? v : -1e9f;
    else
      v = -1e9f;
    Sb[(size_t)m * NP + n] = v;
  }
}

// ---- row softmax -> f16 attention matrix (pad rows/cols = 0) ----
__global__ void k_softmax(const float* __restrict__ Sb, f16* __restrict__ A) {
  __shared__ float red[256];
  int row = blockIdx.x, tid = threadIdx.x;
  if (row >= NCAT) {
    for (int c = tid; c < NP; c += 256) A[(size_t)row * NP + c] = (f16)0.f;
    return;
  }
  const float* s = Sb + (size_t)row * NP;
  float mx = -3.4e38f;
  for (int c = tid; c < NCAT; c += 256) mx = fmaxf(mx, s[c]);
  red[tid] = mx; __syncthreads();
  for (int o = 128; o > 0; o >>= 1) { if (tid < o) red[tid] = fmaxf(red[tid], red[tid + o]); __syncthreads(); }
  mx = red[0]; __syncthreads();
  float sum = 0.f;
  for (int c = tid; c < NCAT; c += 256) sum += __expf(s[c] - mx);
  red[tid] = sum; __syncthreads();
  for (int o = 128; o > 0; o >>= 1) { if (tid < o) red[tid] += red[tid + o]; __syncthreads(); }
  float inv = 1.f / red[0];
  f16* a = A + (size_t)row * NP;
  for (int c = tid; c < NP; c += 256)
    a[c] = (c < NCAT) ? (f16)(__expf(s[c] - mx) * inv) : (f16)0.f;
}

// ---- one diffusion hop: z' = 0.85*(A @ z) + 0.15*v ; last hop also h += z' ----
// Block = 256 threads = 8 waves covering 2 row-tiles x 4 col-tiles; the A
// k-chunk (2KB) is async-staged to LDS once per step and shared by all waves.
__global__ void __launch_bounds__(256)
k_hop(const f16* __restrict__ A, const f16* __restrict__ zC,
      const float* __restrict__ vf, f16* __restrict__ zCout,
      float* __restrict__ hf32, f16* __restrict__ hf16, int writeH) {
  const int tid = threadIdx.x;
  const int wave = tid >> 5, lane = tid & 31;
  const int rtl = wave >> 2, ct = wave & 3;
  const int rt = blockIdx.x * 2 + rtl;
  const int mrow = lane & 15, half = lane >> 4;
  v8f acc = {};

#if HAS_ASYNC_LDS
  __shared__ f16 sA[2][2][16][32];   // [buf][row-tile][row][k] : 4KB
  const int chunk = wave * 16 + lane;          // 8 waves x lanes 0..15 = 128 chunks
  const int crow = chunk >> 2, csub = chunk & 3;
  // prologue: stage k-chunk 0 (one async instr per wave -> uniform ASYNCcnt)
  if (lane < 16)
    async_g2l_b128(A + (size_t)(blockIdx.x * 32 + crow) * NP + csub * 8,
                   (void*)&sA[0][crow >> 4][crow & 15][csub * 8]);
  for (int i = 0; i < NP / 32; ++i) {
    asm volatile("s_wait_asynccnt 0x0" ::: "memory");  // chunk i landed (own wave)
    __syncthreads();                                   // ... for every wave
    if (i + 1 < NP / 32) {                             // overlap next copy w/ compute
      if (lane < 16)
        async_g2l_b128(A + (size_t)(blockIdx.x * 32 + crow) * NP + (i + 1) * 32 + csub * 8,
                       (void*)&sA[(i + 1) & 1][crow >> 4][crow & 15][csub * 8]);
    }
    v16h a = frag_a_rm(&sA[i & 1][rtl][0][0], 32, 0, 0, lane);   // ds_load_b128 x2
    v16h b = frag_b_cm(zC, NP, ct * 16, i * 32, lane);
    acc = WMMA_F32_F16(a, b, acc);
  }
#else
  for (int k0 = 0; k0 < NP; k0 += 32) {
    if (k0 + 32 < NP)  // global_prefetch_b8 the next A-fragment chunk
      __builtin_prefetch(A + (size_t)(rt * 16 + mrow) * NP + k0 + 32 + 8 * half, 0, 1);
    v16h a = frag_a_rm(A, NP, rt * 16, k0, lane);
    v16h b = frag_b_cm(zC, NP, ct * 16, k0, lane);
    acc = WMMA_F32_F16(a, b, acc);
  }
#endif

  int c = ct * 16 + mrow;
  v8h zpack;
#pragma unroll
  for (int r = 0; r < 8; ++r) {
    int m = rt * 16 + 8 * half + r;
    int idx = m * HID + c;
    float z = (1.f - ALPHAC) * acc[r] + ALPHAC * vf[idx];
    zpack[r] = (f16)z;
    if (writeH) {
      float hn = hf32[idx] + z;   // residual
      hf32[idx] = hn;
      hf16[idx] = (f16)hn;
    }
  }
  *(v8h*)(zCout + (size_t)c * NP + rt * 16 + 8 * half) = zpack;
}

// ---- per-node cluster softmax: t[n][g] = softmax_g(h@Wg+bg)*Wp[n], q2[n] = sum_g sm^2 ----
__global__ void k_cluster(const float* __restrict__ h, const float* __restrict__ Wg,
                          const float* __restrict__ bg, const float* __restrict__ Wp,
                          float* __restrict__ t, float* __restrict__ q2) {
  int n = blockIdx.x * 256 + threadIdx.x;
  if (n >= NCAT) return;
  float lg0 = bg[0], lg1 = bg[1], lg2 = bg[2], lg3 = bg[3];
  const float* hr = h + n * HID;
  for (int c = 0; c < HID; ++c) {
    float hv = hr[c];
    lg0 += hv * Wg[c * 4 + 0]; lg1 += hv * Wg[c * 4 + 1];
    lg2 += hv * Wg[c * 4 + 2]; lg3 += hv * Wg[c * 4 + 3];
  }
  float mx = fmaxf(fmaxf(lg0, lg1), fmaxf(lg2, lg3));
  float e0 = __expf(lg0 - mx), e1 = __expf(lg1 - mx);
  float e2 = __expf(lg2 - mx), e3 = __expf(lg3 - mx);
  float inv = 1.f / (e0 + e1 + e2 + e3);
  float s0 = e0 * inv, s1 = e1 * inv, s2 = e2 * inv, s3 = e3 * inv;
  float p = Wp[n];
  t[n * 4 + 0] = s0 * p; t[n * 4 + 1] = s1 * p;
  t[n * 4 + 2] = s2 * p; t[n * 4 + 3] = s3 * p;
  q2[n] = s0 * s0 + s1 * s1 + s2 * s2 + s3 * s3;
}

// ---- S[g][c] = sum_n t[n][g]*h[n][c] ; sgc[256] = sum_n Wp[n] ----
__global__ void k_sgc(const float* __restrict__ t, const float* __restrict__ h,
                      const float* __restrict__ Wp, float* __restrict__ sgc) {
  int tid = threadIdx.x;
  int g = tid >> 6, c = tid & 63;
  int n0 = blockIdx.x * 500;
  int n1 = n0 + 500; if (n1 > NCAT) n1 = NCAT;
  float acc = 0.f;
  for (int n = n0; n < n1; ++n) acc += t[n * 4 + g] * h[n * HID + c];
  atomicAdd(&sgc[g * HID + c], acc);
  if (tid == 0) {
    float w = 0.f;
    for (int n = n0; n < n1; ++n) w += Wp[n];
    atomicAdd(&sgc[NG * HID], w);
  }
}

// ---- per-user mean / inv-std over (G,L): closed form via softmax identities ----
__global__ void k_stats(const int* __restrict__ cate, const float* __restrict__ h,
                        const float* __restrict__ q2, float* __restrict__ meanb,
                        float* __restrict__ istdb) {
  int b = blockIdx.x, c = threadIdx.x;     // 256 blocks x 64 threads
  float s1 = 0.f, s2 = 0.f;
  for (int l = 0; l < LL; ++l) {
    int n = cate[b * LL + l];
    if (n != 0) {
      float hv = h[n * HID + c];
      s1 += hv;                     // sum_g score = 1
      s2 += q2[n] * hv * hv;        // sum_g score^2 = q2
    }
  }
  float mean = s1 * (1.f / (NG * LL));
  float var  = s2 * (1.f / (NG * LL)) - mean * mean;
  meanb[b * HID + c] = mean;
  istdb[b * HID + c] = rsqrtf(var + EPSV);
}

// ---- out[b,g,c] = gamma*istd*(S[g,c] - mean*Wsum) + beta*Wsum + bp ----
__global__ void k_out(const float* __restrict__ sgc, const float* __restrict__ meanb,
                      const float* __restrict__ istdb, const float* __restrict__ gamma,
                      const float* __restrict__ beta, const float* __restrict__ bp,
                      float* __restrict__ out) {
  int b = blockIdx.x, tid = threadIdx.x;
  int g = tid >> 6, c = tid & 63;
  float wsum = sgc[NG * HID];
  float v = gamma[c] * istdb[b * HID + c] * (sgc[g * HID + c] - meanb[b * HID + c] * wsum)
          + beta[c] * wsum + bp[0];
  out[b * NG * HID + g * HID + c] = v;
}

extern "C" void kernel_launch(void* const* d_in, const int* in_sizes, int n_in,
                              void* d_out, int out_size, void* d_ws, size_t ws_size,
                              hipStream_t stream) {
  (void)in_sizes; (void)n_in; (void)out_size; (void)ws_size;  // needs ~58 MB of ws
  const int*   cate  = (const int*)  d_in[0];
  const float* adj   = (const float*)d_in[1];
  const float* emb   = (const float*)d_in[2];
  const float* Wq    = (const float*)d_in[3];
  const float* Wk    = (const float*)d_in[4];
  const float* Wv    = (const float*)d_in[5];
  const float* Wg    = (const float*)d_in[6];
  const float* bg    = (const float*)d_in[7];
  const float* Wp    = (const float*)d_in[8];
  const float* bp    = (const float*)d_in[9];
  const float* gamma = (const float*)d_in[10];
  const float* beta  = (const float*)d_in[11];
  float* out = (float*)d_out;

  char* base = (char*)d_ws;
  size_t off = 0;
  auto carve = [&](size_t bytes) -> char* {
    char* p = base + off;
    off += (bytes + 255) & ~(size_t)255;
    return p;
  };
  float* hf32 = (float*)carve((size_t)NP * HID * 4);
  f16*   hf16 = (f16*)  carve((size_t)NP * HID * 2);
  f16*   qf   = (f16*)  carve((size_t)NP * HID * 2);
  f16*   kf   = (f16*)  carve((size_t)NP * HID * 2);
  float* vf   = (float*)carve((size_t)NP * HID * 4);
  f16*   vC   = (f16*)  carve((size_t)HID * NP * 2);  // column-major v (= z init)
  f16*   z0   = (f16*)  carve((size_t)HID * NP * 2);
  f16*   z1   = (f16*)  carve((size_t)HID * NP * 2);
  f16*   wT   = (f16*)  carve((size_t)3 * NL * HID * HID * 2);
  float* Sb   = (float*)carve((size_t)NP * NP * 4);   // raw masked scores
  f16*   Amat = (f16*)  carve((size_t)NP * NP * 2);   // softmaxed attention
  float* tb   = (float*)carve((size_t)NCAT * 4 * 4);
  float* q2   = (float*)carve((size_t)NCAT * 4);
  float* sgc  = (float*)carve((size_t)(NG * HID + 1) * 4);
  float* meanb= (float*)carve((size_t)BB * HID * 4);
  float* istdb= (float*)carve((size_t)BB * HID * 4);

  f16* wqT = wT;
  f16* wkT = wT + NL * HID * HID;
  f16* wvT = wT + 2 * NL * HID * HID;

  k_init <<<(NP * HID + 255) / 256, 256, 0, stream>>>(emb, hf32, hf16, sgc);
  k_castw<<<(NL * HID * HID + 255) / 256, 256, 0, stream>>>(Wq, Wk, Wv, wqT, wkT, wvT);

  for (int l = 0; l < NL; ++l) {
    k_qkv   <<<94, 256, 0, stream>>>(hf16, wqT + l * HID * HID, wkT + l * HID * HID,
                                     wvT + l * HID * HID, qf, kf, vf, vC);
    k_scores<<<(NT * NT + 7) / 8, 256, 0, stream>>>(qf, kf, adj, Sb);
    k_softmax<<<NP, 256, 0, stream>>>(Sb, Amat);
    const f16* zi = vC;
    f16* zo = z0;
    for (int hop = 0; hop < HOPS; ++hop) {
      int last = (hop == HOPS - 1) ? 1 : 0;
      k_hop<<<94, 256, 0, stream>>>(Amat, zi, vf, zo, hf32, hf16, last);
      zi = zo;
      zo = (zo == z0) ? z1 : z0;
    }
  }

  k_cluster<<<(NCAT + 255) / 256, 256, 0, stream>>>(hf32, Wg, bg, Wp, tb, q2);
  k_sgc    <<<6, 256, 0, stream>>>(tb, hf32, Wp, sgc);
  k_stats  <<<BB, 64, 0, stream>>>(cate, hf32, q2, meanb, istdb);
  k_out    <<<BB, 256, 0, stream>>>(sgc, meanb, istdb, gamma, beta, bp, out);
}
